// InvertibleBigraphNeuralNetwork_7576322310151
// MI455X (gfx1250) — matrix operand order repaired and checked by reading, not compiled, not run on hardware
//
#include <hip/hip_runtime.h>

#define BB 4096
#define NN 41
#define DD 271
#define MROWS (BB*NN)        // 167936
#define KP 288               // K padded (9 * 32)
#define NP 272               // N padded (17 * 16)
#define KTILES 9
#define NTILES 17
#define MTILES (MROWS/16)    // 10496
#define NCHUNKS 5            // 4 chunks of 4 N-tiles + 1 remainder tile

typedef __attribute__((ext_vector_type(16))) __bf16        bf16x16;
typedef __attribute__((ext_vector_type(8)))  float         f32x8;
typedef __attribute__((ext_vector_type(4)))  unsigned int  u32x4;

union Frag { u32x4 q[2]; bf16x16 v; };

__device__ __forceinline__ unsigned short f2bf(float f){
  unsigned int u = __builtin_bit_cast(unsigned int, f);
  unsigned int r = (u + 0x7FFFu + ((u >> 16) & 1u)) >> 16;
  return (unsigned short)r;
}

// ---------- weight pre-pack into WMMA B-fragment layout ----------
// packed[((kt*17+nt)*32+lane)*8 + v] = {B[k0+2v], B[k0+2v+1]} at col nt*16+(lane&15),
// k0 = kt*32 + (lane>=16 ? 16 : 0)
__global__ void pack_b_kernel(const float* __restrict__ s0, const float* __restrict__ s1,
                              const float* __restrict__ s2, unsigned int* __restrict__ d0,
                              unsigned int* __restrict__ d1, unsigned int* __restrict__ d2){
  int t = blockIdx.x*blockDim.x + threadIdx.x;
  const int total = KTILES*NTILES*32*8;
  if (t >= total) return;
  const float* src = (blockIdx.y==0) ? s0 : (blockIdx.y==1 ? s1 : s2);
  unsigned int* dst = (blockIdx.y==0) ? d0 : (blockIdx.y==1 ? d1 : d2);
  int v    = t & 7;
  int lane = (t >> 3) & 31;
  int nt   = (t >> 8) % NTILES;
  int kt   = (t >> 8) / NTILES;
  int col  = nt*16 + (lane & 15);
  int k0   = kt*32 + ((lane >> 4) ? 16 : 0) + 2*v;
  float lo = (k0   < DD && col < DD) ? src[k0*DD + col]     : 0.f;
  float hi = (k0+1 < DD && col < DD) ? src[(k0+1)*DD + col] : 0.f;
  dst[t] = (unsigned int)f2bf(lo) | ((unsigned int)f2bf(hi) << 16);
}

// ---------- X fp32 -> bf16, K-padded to 288 ----------
__global__ void conv_x_kernel(const float* __restrict__ X, unsigned short* __restrict__ Xb){
  long long idx = (long long)blockIdx.x*blockDim.x + threadIdx.x;
  if (idx >= (long long)MROWS*KP) return;
  int c = (int)(idx % KP);
  long long row = idx / KP;
  float val = (c < DD) ? X[row*DD + c] : 0.f;
  Xb[idx] = f2bf(val);
}

__global__ void coff_kernel(const float* __restrict__ lsv, float* __restrict__ coff){
  int i = blockIdx.x*blockDim.x + threadIdx.x;
  if (i < NN*DD) coff[i] = expf(lsv[i]);
}

// ---------- per-sample wavelet basis ----------
__global__ void wavelet_kernel(const float* __restrict__ G, float* __restrict__ Wv,
                               float* __restrict__ WiM){
  __shared__ float sG[NN*NN], sL[NN*NN], sT2[NN*NN], sW[NN*NN], sWi[NN*NN];
  __shared__ float deg[NN], dinv[NN], pv[NN], pw[NN], rs[2*NN], sc[4], sic[4], slam;
  int b = blockIdx.x, tid = threadIdx.x;
  const float* g = G + (long long)b*NN*NN;
  for (int i=tid;i<NN*NN;i+=blockDim.x) sG[i] = g[i];
  __syncthreads();
  if (tid < NN){
    float s=0.f;
    for (int j=0;j<NN;j++) s += sG[tid*NN+j];
    deg[tid]=s; dinv[tid]=1.f/(sqrtf(s)+1e-6f); pv[tid]=1.f+0.01f*(float)tid;
  }
  __syncthreads();
  for (int e=tid;e<NN*NN;e+=blockDim.x){ int i=e/NN, j=e%NN;
    float a = 0.5f*(sG[i*NN+j]+sG[j*NN+i]);
    sL[e] = dinv[i]*(((i==j)?deg[i]:0.f) - a)*dinv[j];
  }
  __syncthreads();
  for (int e=tid;e<NN*NN;e+=blockDim.x){ int i=e/NN, j=e%NN;
    float s=0.f;
    for (int k=0;k<NN;k++) s += sL[i*NN+k]*sL[k*NN+j];
    sT2[e] = 2.f*s;
  }
  __syncthreads();
  for (int e=tid;e<NN*NN;e+=blockDim.x){ int i=e/NN, j=e%NN;
    float s=0.f;
    for (int k=0;k<NN;k++) s += sL[i*NN+k]*sT2[k*NN+j];
    sG[e] = 2.f*s - sL[e];              // T3 overwrites sG (G no longer needed)
  }
  __syncthreads();
  // power iteration for lambda_max (L symmetric PSD)
  for (int it=0; it<48; ++it){
    if (tid < NN){ float s=0.f; for (int j=0;j<NN;j++) s += sL[tid*NN+j]*pv[j]; pw[tid]=s; }
    __syncthreads();
    if (tid == 0){ float s=0.f; for (int j=0;j<NN;j++) s += pw[j]*pw[j]; slam = sqrtf(s); }
    __syncthreads();
    if (tid < NN) pv[tid] = pw[tid] / fmaxf(slam, 1e-20f);
    __syncthreads();
  }
  if (tid == 0){
    float lam = slam;
    float x   = 0.3f*0.5f*lam;          // s * a
    float xh  = 0.5f*x, xh2 = xh*xh;
    float ex  = expf(x), exn = expf(-x);
    for (int n=0;n<4;n++){
      float term = 1.f;
      for (int m=1;m<=n;m++) term *= xh/(float)m;     // xh^n / n!
      float s = 0.f;
      for (int k=0;k<25;k++){ s += term; term *= xh2/((float)(k+1)*(float)(k+1+n)); }
      sc[n]  = 2.f*ex *s;
      sic[n] = 2.f*exn*s*((n&1)?-1.f:1.f);
    }
  }
  __syncthreads();
  for (int e=tid;e<NN*NN;e+=blockDim.x){
    float w  = sc[1]*sL[e]  + sc[2]*sT2[e]  + sc[3]*sG[e];
    float wi = sic[1]*sL[e] + sic[2]*sT2[e] + sic[3]*sG[e];
    sW[e]  = (w  < 1e-4f) ? 0.f : w;
    sWi[e] = (wi < 1e-4f) ? 0.f : wi;
  }
  __syncthreads();
  if (tid < 2*NN){
    int i = tid % NN;
    const float* src = (tid < NN) ? sW : sWi;
    float s=0.f;
    for (int j=0;j<NN;j++) s += fabsf(src[i*NN+j]);
    rs[tid] = 1.f / fmaxf(s, 1e-12f);
  }
  __syncthreads();
  for (int e=tid;e<NN*NN;e+=blockDim.x){
    int i = e/NN;
    Wv [(long long)b*NN*NN+e] = sW[e] *rs[i];
    WiM[(long long)b*NN*NN+e] = sWi[e]*rs[NN+i];
  }
}

// ---------- bf16 WMMA GEMM: [MROWS,288]bf16 x packed[288,272]bf16 -> out ----------
// 2x4 register tiling: each wave computes 32 rows x (16*NV) cols.
// MODE 0: f32 out (ld=272), no bias; MODE 1: bias+relu, bf16 out (ld=288, pad zeroed);
// MODE 2: bias, f32 out (ld=272)
template<int MODE, int NV>
__device__ __forceinline__ void gemm_body(int mt2, int nt0, int lane,
                                          const unsigned short* __restrict__ A,
                                          const unsigned int* __restrict__ PB,
                                          const float* __restrict__ bias,
                                          void* __restrict__ out){
  int hi = lane >> 4;
  int r0 = mt2*32 + (lane & 15);
  const unsigned short* arow0 = A + (long long)r0*KP + hi*16;
  const unsigned short* arow1 = arow0 + (long long)16*KP;
  const unsigned int*   pbl   = PB + ((size_t)(nt0*32 + lane))*8;

  f32x8 acc[2][NV] = {};

  #pragma unroll
  for (int kt=0; kt<KTILES; ++kt){
    // load all fragments for this K-step first (distinct registers -> one big clause)
    Frag a0, a1, b[NV];
    {
      const u32x4* pa0 = (const u32x4*)(arow0 + kt*32);
      a0.q[0] = pa0[0]; a0.q[1] = pa0[1];     // K = k0..k0+15 contiguous (k0 = 16*half-wave)
      const u32x4* pa1 = (const u32x4*)(arow1 + kt*32);
      a1.q[0] = pa1[0]; a1.q[1] = pa1[1];
    }
    #pragma unroll
    for (int j=0;j<NV;j++){
      const u32x4* pb = (const u32x4*)(pbl + (size_t)kt*(NTILES*256) + j*256);
      b[j].q[0] = pb[0]; b[j].q[1] = pb[1];
    }
    // then the WMMA burst
    #pragma unroll
    for (int j=0;j<NV;j++){
      acc[0][j] = __builtin_amdgcn_wmma_f32_16x16x32_bf16(
                    false, a0.v, false, b[j].v, (short)0, acc[0][j], false, false);
      acc[1][j] = __builtin_amdgcn_wmma_f32_16x16x32_bf16(
                    false, a1.v, false, b[j].v, (short)0, acc[1][j], false, false);
    }
  }

  #pragma unroll
  for (int m=0;m<2;m++){
    #pragma unroll
    for (int j=0;j<NV;j++){
      int nt  = nt0 + j;
      int col = nt*16 + (lane & 15);
      float bv = 0.f;
      if (MODE != 0 && col < DD) bv = bias[col];
      #pragma unroll
      for (int i=0;i<8;i++){
        int r = mt2*32 + m*16 + i + hi*8;
        float val = acc[m][j][i] + bv;
        if (MODE == 1){
          val = fmaxf(val, 0.f);
          ((unsigned short*)out)[(long long)r*KP + col] = f2bf(val);
        } else {
          ((float*)out)[(long long)r*NP + col] = val;
        }
      }
      if (MODE == 1 && nt == NTILES-1){
        // zero K-pad columns 272..287 for this 16-row tile
        int rr = lane >> 1;
        int cc = NP + (lane & 1)*8;
        u32x4 z = {};
        *(u32x4*)((unsigned short*)out + (long long)(mt2*32 + m*16 + rr)*KP + cc) = z;
      }
    }
  }
}

template<int MODE>
__global__ void gemm_kernel(const unsigned short* __restrict__ A,
                            const unsigned int* __restrict__ PB,
                            const float* __restrict__ bias,
                            void* __restrict__ out){
  int wave = blockIdx.x*(blockDim.x >> 5) + (threadIdx.x >> 5);
  int lane = threadIdx.x & 31;
  // wave-uniform tile coordinates: force into SGPRs so branches are scalar
  int mt2   = __builtin_amdgcn_readfirstlane(wave / NCHUNKS);
  int chunk = __builtin_amdgcn_readfirstlane(wave % NCHUNKS);
  if (chunk == NCHUNKS-1)
    gemm_body<MODE,1>(mt2, (NCHUNKS-1)*4, lane, A, PB, bias, out);
  else
    gemm_body<MODE,4>(mt2, chunk*4, lane, A, PB, bias, out);
}

// ---------- fused wav_inv @ h, diag(kernel), relu(wav @ .), coff scale, ->bf16 ----------
__global__ void middle_kernel(const float* __restrict__ Wv, const float* __restrict__ WiM,
                              const float* __restrict__ h1, const float* __restrict__ kern,
                              const float* __restrict__ coff, unsigned short* __restrict__ A2){
  __shared__ float swv[NN*NN], swi[NN*NN], skern[NN], tin[NN*96], su[NN*96];
  int b = blockIdx.x, tid = threadIdx.x;
  for (int i=tid;i<NN*NN;i+=blockDim.x){
    swv[i] = Wv [(long long)b*NN*NN+i];
    swi[i] = WiM[(long long)b*NN*NN+i];
  }
  if (tid < NN) skern[tid] = kern[tid];
  // zero K-pad cols 271..287 of A2 for this sample's rows
  for (int i=tid;i<NN*17;i+=blockDim.x){
    int r = i/17, c = DD + i%17;
    A2[((long long)b*NN+r)*KP + c] = 0;
  }
  for (int c0=0; c0<DD; c0+=96){
    int w = (DD - c0 < 96) ? (DD - c0) : 96;
    __syncthreads();
    for (int e=tid;e<NN*w;e+=blockDim.x){
      int j=e/w, cc=e%w;
      tin[j*96+cc] = h1[((long long)b*NN+j)*NP + c0+cc];
    }
    __syncthreads();
    for (int e=tid;e<NN*w;e+=blockDim.x){
      int i=e/w, cc=e%w;
      float s=0.f;
      for (int j=0;j<NN;j++) s += swi[i*NN+j]*tin[j*96+cc];
      su[i*96+cc] = skern[i]*s;
    }
    __syncthreads();
    for (int e=tid;e<NN*w;e+=blockDim.x){
      int i=e/w, cc=e%w;
      float s=0.f;
      for (int j=0;j<NN;j++) s += swv[i*NN+j]*su[j*96+cc];
      float fr = coff[i*DD + c0+cc] * fmaxf(s, 0.f);
      A2[((long long)b*NN+i)*KP + c0+cc] = f2bf(fr);
    }
  }
}

// ---------- mean over nodes ----------
__global__ void seq_kernel(const float* __restrict__ g2, float* __restrict__ seq){
  int idx = blockIdx.x*blockDim.x + threadIdx.x;
  if (idx >= BB*DD) return;
  int b = idx / DD, d = idx % DD;
  float s=0.f;
  for (int n=0;n<NN;n++) s += g2[((long long)b*NN+n)*NP + d];
  seq[idx] = s * (1.f/(float)NN);
}

// ---------- BN stats (deterministic tree reduce per feature) ----------
__global__ void bnstat_kernel(const float* __restrict__ seq, float* __restrict__ sums){
  __shared__ float s1[256], s2[256];
  int d = blockIdx.x, tid = threadIdx.x;
  float a=0.f, q=0.f;
  for (int b=tid;b<BB;b+=256){ float v = seq[b*DD+d]; a += v; q += v*v; }
  s1[tid]=a; s2[tid]=q; __syncthreads();
  for (int st=128; st>0; st>>=1){
    if (tid<st){ s1[tid]+=s1[tid+st]; s2[tid]+=s2[tid+st]; }
    __syncthreads();
  }
  if (tid==0){ sums[d]=s1[0]; sums[DD+d]=s2[0]; }
}

__global__ void bn_kernel(const float* __restrict__ seq, const float* __restrict__ sums,
                          const float* __restrict__ gamma, const float* __restrict__ beta,
                          float* __restrict__ out){
  int idx = blockIdx.x*blockDim.x + threadIdx.x;
  if (idx >= BB*DD) return;
  int d = idx % DD;
  float m = sums[d]    * (1.f/(float)BB);
  float v = sums[DD+d] * (1.f/(float)BB) - m*m;
  out[idx] = (seq[idx]-m)*rsqrtf(v+1e-5f)*gamma[d] + beta[d];
}

extern "C" void kernel_launch(void* const* d_in, const int* in_sizes, int n_in,
                              void* d_out, int out_size, void* d_ws, size_t ws_size,
                              hipStream_t stream){
  const float* G   = (const float*)d_in[0];
  const float* X   = (const float*)d_in[1];
  const float* gk  = (const float*)d_in[2];
  const float* gW  = (const float*)d_in[3];
  const float* lsv = (const float*)d_in[4];
  const float* W1  = (const float*)d_in[5];
  const float* b1  = (const float*)d_in[6];
  const float* W2  = (const float*)d_in[7];
  const float* b2  = (const float*)d_in[8];
  const float* bg  = (const float*)d_in[9];
  const float* bbt = (const float*)d_in[10];
  float* out = (float*)d_out;
  (void)in_sizes; (void)n_in; (void)out_size; (void)ws_size;

  char* ws = (char*)d_ws;
  size_t off = 0;
  auto alloc = [&](size_t bytes)->char*{
    char* p = ws + off;
    off = (off + bytes + 255) & ~(size_t)255;
    return p;
  };
  const size_t PBDW = (size_t)KTILES*NTILES*32*8;                  // dwords per packed weight
  unsigned int*   pb0  = (unsigned int*)  alloc(PBDW*4);
  unsigned int*   pb1  = (unsigned int*)  alloc(PBDW*4);
  unsigned int*   pb2  = (unsigned int*)  alloc(PBDW*4);
  float*          coff = (float*)         alloc((size_t)NN*DD*4);
  float*          wavW = (float*)         alloc((size_t)BB*NN*NN*4);
  float*          wavI = (float*)         alloc((size_t)BB*NN*NN*4);
  unsigned short* Xbf  = (unsigned short*)alloc((size_t)MROWS*KP*2);   // also A3
  float*          h1   = (float*)         alloc((size_t)MROWS*NP*4);   // also g2
  unsigned short* A2   = (unsigned short*)alloc((size_t)MROWS*KP*2);
  float*          seq  = (float*)         alloc((size_t)BB*DD*4);
  float*          sums = (float*)         alloc((size_t)2*DD*4);

  dim3 pgrid((unsigned)((PBDW + 255)/256), 3);
  pack_b_kernel<<<pgrid, 256, 0, stream>>>(gW, W1, W2, pb0, pb1, pb2);
  coff_kernel<<<(NN*DD + 255)/256, 256, 0, stream>>>(lsv, coff);
  conv_x_kernel<<<(unsigned)(((long long)MROWS*KP + 255)/256), 256, 0, stream>>>(X, Xbf);
  wavelet_kernel<<<BB, 128, 0, stream>>>(G, wavW, wavI);

  const int GEMM_BLOCKS = ((MTILES/2)*NCHUNKS)/8;   // 3280 blocks * 8 waves = 26240 waves
  gemm_kernel<0><<<GEMM_BLOCKS, 256, 0, stream>>>(Xbf, pb0, nullptr, (void*)h1);
  middle_kernel<<<BB, 256, 0, stream>>>(wavW, wavI, h1, gk, coff, A2);
  gemm_kernel<1><<<GEMM_BLOCKS, 256, 0, stream>>>(A2,  pb1, b1, (void*)Xbf); // A3 aliases Xbf
  gemm_kernel<2><<<GEMM_BLOCKS, 256, 0, stream>>>(Xbf, pb2, b2, (void*)h1);  // g2 aliases h1

  seq_kernel<<<(BB*DD + 255)/256, 256, 0, stream>>>(h1, seq);
  bnstat_kernel<<<DD, 256, 0, stream>>>(seq, sums);
  bn_kernel<<<(BB*DD + 255)/256, 256, 0, stream>>>(seq, sums, bg, bbt, out);
}
